// RandLANET_54193897341159
// MI455X (gfx1250) — matrix operand order repaired and checked by reading, not compiled, not run on hardware
//
#include <hip/hip_runtime.h>
#include <math.h>

typedef _Float16 f16_t;
typedef __attribute__((ext_vector_type(16))) _Float16 v16h;
typedef __attribute__((ext_vector_type(8)))  float    v8f;

__device__ __forceinline__ float lrelu(float v) { return v > 0.0f ? v : 0.2f * v; }

// ---------------------------------------------------------------------------
// Generic 1x1 conv (pointwise GEMM) + folded BN + LeakyReLU via
// V_WMMA_F32_16X16X32_F16 (f16 inputs, f32 accumulate).
// Input channel c: c < C1 from X1 [B,C1,NT]; C1 <= c < C1+C2 from X2 [B,C2,NT]
// (implements jnp.concatenate along channels without materializing it).
// Grid: x = n-tile group (8 waves/block), y = 16-wide out-channel tile,
// z = batch  ->  zero integer div/mod on the wave's critical path.
//
// EXACT=true requires Cout%16==0, C1%32==0 and (C2==0 || C2%32==0): every
// 32-channel K-chunk then comes entirely from X1 or entirely from X2, so the
// source select is a wave-uniform per-chunk branch and the unrolled fragment
// loop is pure load+cvt (no per-element compares).  Used for all the
// FLOP-dominant layers; the generic path covers fc0/enc0/fc3 shapes.
template <bool EXACT>
__global__ void pw_conv_wmma(const float* __restrict__ X1, int C1,
                             const float* __restrict__ X2, int C2,
                             const float* __restrict__ W,
                             const float* __restrict__ bs,
                             const float* __restrict__ bb,
                             float* __restrict__ Y,
                             int Cout, int NT, int act)
{
  const int tilesN = NT >> 4;
  const int nt = blockIdx.x * 8 + (threadIdx.x >> 5);
  if (nt >= tilesN) return;                  // wave-uniform: EXEC stays all-1s
  const int Cin = C1 + C2;
  const int lane = threadIdx.x & 31;
  const int rc = lane & 15;                  // A row / B column
  const int hi = lane >> 4;
  const int o0 = blockIdx.y << 4;
  const int b  = blockIdx.z;
  const int n  = (nt << 4) + rc;
  const float* wrow = W + (o0 + rc) * Cin;   // A row for this lane

  v8f acc = {};
  const int kSteps = (Cin + 31) >> 5;
  for (int ks = 0; ks < kSteps; ++ks) {
    const int c0 = ks << 5;
    v16h af, bf;
    if (EXACT) {
      const bool useX1 = c0 < C1;            // chunk never straddles the split
      const float* xb = useX1 ? X1 + ((size_t)(b * C1 + c0) + (size_t)(hi << 4)) * NT + n
                              : X2 + ((size_t)(b * C2 + (c0 - C1)) + (size_t)(hi << 4)) * NT + n;
#pragma unroll
      for (int e = 0; e < 16; ++e) {
        const int ka = e + (hi << 4);
        af[e] = (f16_t)wrow[c0 + ka];
        bf[e] = (f16_t)xb[(size_t)e * NT];
      }
    } else {
#pragma unroll
      for (int e = 0; e < 16; ++e) {
        // A-fragment K index: K = e + 16*hi (matches B)
        const int ka = e + (hi << 4);
        const int ca = c0 + ka;
        af[e] = (f16_t)((ca < Cin && (o0 + rc) < Cout) ? wrow[ca] : 0.0f);
        // B-fragment K index: K = e + 16*hi, col = rc
        const int cb = c0 + e + (hi << 4);
        float xv = 0.0f;
        if (cb < C1)        xv = X1[(b * C1 + cb) * NT + n];
        else if (cb < Cin)  xv = X2[(b * C2 + (cb - C1)) * NT + n];
        bf[e] = (f16_t)xv;
      }
    }
    acc = __builtin_amdgcn_wmma_f32_16x16x32_f16(false, af, false, bf,
                                                 (short)0, acc, false, false);
  }
#pragma unroll
  for (int r = 0; r < 8; ++r) {
    const int o = o0 + r + (hi << 3);        // D row = r + 8*hi
    if (EXACT || o < Cout) {
      float v = acc[r];
      if (bs)  v = v * bs[o] + bb[o];
      if (act) v = lrelu(v);
      Y[(b * Cout + o) * NT + n] = v;
    }
  }
}

// ---------------------------------------------------------------------------
// Fused attention pooling:
//   f = Wfc * concat(gather(G, neigh), FX)   (raw einsum, no BN/act)
//   sc = softmax(f, axis=K); ws = sum_k f*sc
// K == 16, so one 16x16 WMMA tile = 16 channels x the 16 neighbors of one
// point; softmax/weighted-sum is a 16-lane __shfl_xor reduction in the
// epilogue -> the [B,co,N,K] fc tensor never touches memory.
// Grid: x = point group (8 waves/block), y = out-channel tile, z = batch.
// EXACT=true requires C1%32==0 (co>=64): each 32-channel chunk is then all
// gather or all dense (wave-uniform select), no per-element compares.
template <bool EXACT>
__global__ void attpool_wmma(const float* __restrict__ G, int C1,   // [B,C1,N]
                             const float* __restrict__ FX, int C2,  // [B,C2,N*16]
                             const int* __restrict__ neigh,         // [B,N,16]
                             const float* __restrict__ W,           // [co,co]
                             float* __restrict__ WS,                // [B,co,N]
                             int N)
{
  const int n = blockIdx.x * 8 + (threadIdx.x >> 5);
  if (n >= N) return;
  const int Cin = C1 + C2;                   // square fc: Cout == Cin == co
  const int lane = threadIdx.x & 31;
  const int rc = lane & 15;
  const int hi = lane >> 4;
  const int o0 = blockIdx.y << 4;
  const int b  = blockIdx.z;
  const int k  = rc;                         // neighbor slot = tile column
  const int idx = neigh[(b * N + n) * 16 + k];
  const int NT  = N << 4;
  const int col = (n << 4) + k;
  const float* wrow = W + (o0 + rc) * Cin;

  v8f acc = {};
  const int kSteps = (Cin + 31) >> 5;
  for (int ks = 0; ks < kSteps; ++ks) {
    const int c0 = ks << 5;
    v16h af, bf;
    if (EXACT) {
      const bool gat = c0 < C1;
      const float* xb = gat ? G + ((size_t)(b * C1 + c0) + (size_t)(hi << 4)) * N + idx
                            : FX + ((size_t)(b * C2 + (c0 - C1)) + (size_t)(hi << 4)) * NT + col;
      const size_t stride = gat ? (size_t)N : (size_t)NT;
#pragma unroll
      for (int e = 0; e < 16; ++e) {
        const int ka = e + (hi << 4);
        af[e] = (f16_t)wrow[c0 + ka];
        bf[e] = (f16_t)xb[(size_t)e * stride];
      }
    } else {
#pragma unroll
      for (int e = 0; e < 16; ++e) {
        const int ka = e + (hi << 4);
        const int ca = c0 + ka;
        af[e] = (f16_t)((ca < Cin) ? wrow[ca] : 0.0f);
        const int cb = c0 + e + (hi << 4);
        float xv = 0.0f;
        if (cb < C1)       xv = G[(b * C1 + cb) * N + idx];          // fused gather
        else if (cb < Cin) xv = FX[(b * C2 + (cb - C1)) * NT + col]; // dense half
        bf[e] = (f16_t)xv;
      }
    }
    acc = __builtin_amdgcn_wmma_f32_16x16x32_f16(false, af, false, bf,
                                                 (short)0, acc, false, false);
  }
#pragma unroll
  for (int r = 0; r < 8; ++r) {
    float f = acc[r];
    float m = f;
#pragma unroll
    for (int d = 1; d < 16; d <<= 1) m = fmaxf(m, __shfl_xor(m, d, 32));
    const float e  = __expf(f - m);
    float se = e, sfe = f * e;
#pragma unroll
    for (int d = 1; d < 16; d <<= 1) {
      se  += __shfl_xor(se,  d, 32);
      sfe += __shfl_xor(sfe, d, 32);
    }
    if (k == 0)
      WS[(b * Cin + o0 + r + (hi << 3)) * N + n] = sfe / se;
  }
}

// ---------------------------------------------------------------------------
// bbmlp1 fused with geometric feature construction: each lane owns one (n,k)
// column and builds its 10 features [dist, rel, tile, nb] in registers, then
// feeds them straight into the WMMA B fragment (Cin=10 -> single K chunk).
__global__ void geo_mlp_wmma(const float* __restrict__ xyz,   // [B,N,3]
                             const int* __restrict__ neigh,   // [B,N,16]
                             const float* __restrict__ W,     // [co2,10]
                             const float* __restrict__ bs,
                             const float* __restrict__ bb,
                             float* __restrict__ FXo,         // [B,co2,N*16]
                             int co2, int N)
{
  const int n = blockIdx.x * 8 + (threadIdx.x >> 5);
  if (n >= N) return;
  const int lane = threadIdx.x & 31;
  const int rc = lane & 15;
  const int hi = lane >> 4;
  const int o0 = blockIdx.y << 4;
  const int b  = blockIdx.z;
  const int k  = rc;
  const int idx = neigh[(b * N + n) * 16 + k];
  const float tx = xyz[(b * N + n) * 3 + 0];
  const float ty = xyz[(b * N + n) * 3 + 1];
  const float tz = xyz[(b * N + n) * 3 + 2];
  const float nx = xyz[(b * N + idx) * 3 + 0];
  const float ny = xyz[(b * N + idx) * 3 + 1];
  const float nz = xyz[(b * N + idx) * 3 + 2];
  const float rx = tx - nx, ry = ty - ny, rz = tz - nz;
  const float ds = sqrtf(rx * rx + ry * ry + rz * rz);
  const float geo[10] = {ds, rx, ry, rz, tx, ty, tz, nx, ny, nz};

  v16h af, bf;
#pragma unroll
  for (int e = 0; e < 16; ++e) {
    const int ka = e + (hi << 4);
    af[e] = (f16_t)((ka < 10 && (o0 + rc) < co2) ? W[(o0 + rc) * 10 + ka] : 0.0f);
    float xv = 0.0f;
    if (e < 10) xv = (hi == 0) ? geo[e] : 0.0f;  // B rows 10..31 are zero pad
    bf[e] = (f16_t)xv;
  }
  v8f acc = {};
  acc = __builtin_amdgcn_wmma_f32_16x16x32_f16(false, af, false, bf,
                                               (short)0, acc, false, false);
  const int NT  = N << 4;
  const int col = (n << 4) + k;
#pragma unroll
  for (int r = 0; r < 8; ++r) {
    const int o = o0 + r + (hi << 3);
    if (o < co2)
      FXo[(b * co2 + o) * NT + col] = lrelu(acc[r] * bs[o] + bb[o]);
  }
}

// ---------------------------------------------------------------------------
__global__ void add_lrelu_k(const float* __restrict__ A,
                            const float* __restrict__ Bv,
                            float* __restrict__ O, int n)
{
  int i = blockIdx.x * blockDim.x + threadIdx.x;
  if (i < n) O[i] = lrelu(A[i] + Bv[i]);
}

// Grid: x = point, y = channel, z = batch (no div/mod).
__global__ void sample_max_k(const float* __restrict__ X,  // [B,d,N]
                             const int* __restrict__ sub,  // [B,M,16]
                             float* __restrict__ O,        // [B,d,M]
                             int d, int N, int M)
{
  const int m = blockIdx.x * blockDim.x + threadIdx.x;
  if (m >= M) return;
  const int c = blockIdx.y;
  const int b = blockIdx.z;
  const int* s = &sub[(b * M + m) * 16];
  const float* x = &X[(b * d + c) * N];
  float v = -3.0e38f;
#pragma unroll
  for (int k = 0; k < 16; ++k) v = fmaxf(v, x[s[k]]);
  O[(b * d + c) * M + m] = v;
}

__global__ void interp_k(const float* __restrict__ X,   // [B,d,Nq]
                         const int* __restrict__ idx,   // [B,M,1]
                         float* __restrict__ O,         // [B,d,M]
                         int d, int Nq, int M)
{
  const int m = blockIdx.x * blockDim.x + threadIdx.x;
  if (m >= M) return;
  const int c = blockIdx.y;
  const int b = blockIdx.z;
  O[(b * d + c) * M + m] = X[(b * d + c) * Nq + idx[b * M + m]];
}

// ---------------------------------------------------------------------------
extern "C" void kernel_launch(void* const* d_in, const int* in_sizes, int n_in,
                              void* d_out, int out_size, void* d_ws, size_t ws_size,
                              hipStream_t stream)
{
  (void)n_in; (void)out_size; (void)ws_size;
  const int B = 4;
  const int NS[5] = {40960, 10240, 2560, 640, 160};
  const int DIMS[4] = {16, 64, 128, 256};

  // setup_inputs() dict order: 117 flat param arrays, then 17 data arrays.
  // Auto-detect params-first vs data-first via entry-0 size (fc0_W == 24).
  int pbase = 0, ibase = 117;
  if (in_sizes[0] > 1024) { ibase = 0; pbase = 17; }
  auto P = [&](int i) { return (const float*)d_in[pbase + i]; };
  auto I = [&](int i) { return d_in[ibase + i]; };

  // -------- workspace layout (fp32 elements, ~366 MB total) --------
  float* ws = (float*)d_ws;
  size_t off = 0;
  auto alloc = [&](size_t nel) {
    float* p = ws + off; off += (nel + 63) & ~(size_t)63; return p;
  };
  float* FX1 = alloc(20971520);            // [B,co/2,N*K] max (enc0/enc1)
  float* FX2 = alloc(20971520);
  float* FS[5];
  FS[0] = alloc((size_t)B * 8 * NS[0]);    // fc0 output f0
  FS[1] = alloc((size_t)B * 32 * NS[1]);   // sampled features (enc list)
  FS[2] = alloc((size_t)B * 128 * NS[2]);
  FS[3] = alloc((size_t)B * 256 * NS[3]);
  FS[4] = alloc((size_t)B * 512 * NS[4]);
  float* ENC0 = alloc((size_t)B * 32 * NS[0]);  // fe0 (persistent for decoder)
  float* S1  = alloc(1310720);             // mlp1 output
  float* WSb = alloc(2621440);             // attention-pool output
  float* AGG = alloc(1310720);             // ap1 mlp output
  float* BBo = alloc(2621440);             // ap2 mlp output
  float* M2  = alloc(5242880);             // mlp2 output / fe scratch
  float* M3  = alloc(5242880);             // mlp3 shortcut
  float* FI  = alloc(5242880);             // decoder interp output
  float* PP1 = alloc(10485760);            // decoder ping
  float* PP2 = alloc(5242880);             // decoder pong

  auto pw = [&](const float* x1, int c1, const float* x2, int c2,
                const float* w, const float* s, const float* bb2,
                float* y, int cout, int nt, int act) {
    const int tilesO = (cout + 15) >> 4, tilesN = nt >> 4;
    dim3 grid((tilesN + 7) / 8, tilesO, B);
    const bool exact = (cout % 16 == 0) && (c1 % 32 == 0) &&
                       (c2 == 0 || c2 % 32 == 0);
    if (exact)
      pw_conv_wmma<true><<<grid, 256, 0, stream>>>(x1, c1, x2, c2, w, s, bb2,
                                                   y, cout, nt, act);
    else
      pw_conv_wmma<false><<<grid, 256, 0, stream>>>(x1, c1, x2, c2, w, s, bb2,
                                                    y, cout, nt, act);
  };
  auto ap = [&](const float* g, int c1, const float* fx, int c2,
                const int* neigh, const float* w, float* wsout, int N) {
    dim3 grid((N + 7) / 8, (c1 + c2) >> 4, B);
    if (c1 % 32 == 0 && c2 % 32 == 0)
      attpool_wmma<true><<<grid, 256, 0, stream>>>(g, c1, fx, c2, neigh, w,
                                                   wsout, N);
    else
      attpool_wmma<false><<<grid, 256, 0, stream>>>(g, c1, fx, c2, neigh, w,
                                                    wsout, N);
  };

  // fc0: features [B,3,N0] -> f0 [B,8,N0]
  pw((const float*)I(0), 3, nullptr, 0, P(0), P(1), P(2), FS[0], 8, NS[0], 1);

  const float* fin = FS[0];
  int cin = 8;
  for (int i = 0; i < 4; ++i) {
    const int N = NS[i], co = DIMS[i], co2 = co / 2, NT = N * 16;
    const int pi = 3 + 23 * i;
    const float* xyz   = (const float*)I(1 + 4 * i);
    const int*   neigh = (const int*)I(2 + 4 * i);
    const int*   sub   = (const int*)I(3 + 4 * i);

    // mlp1: fin -> S1 [B,co2,N]
    pw(fin, cin, nullptr, 0, P(pi + 0), P(pi + 1), P(pi + 2), S1, co2, N, 1);
    // bbmlp1 (+on-the-fly geometry) -> FX1 [B,co2,NT]
    {
      dim3 grid((N + 7) / 8, (co2 + 15) >> 4, B);
      geo_mlp_wmma<<<grid, 256, 0, stream>>>(
          xyz, neigh, P(pi + 3), P(pi + 4), P(pi + 5), FX1, co2, N);
    }
    // ap1 fc + softmax(K) + weighted sum -> WSb [B,co,N]
    ap(S1, co2, FX1, co2, neigh, P(pi + 6), WSb, N);
    // ap1 mlp -> AGG [B,co2,N]
    pw(WSb, co, nullptr, 0, P(pi + 7), P(pi + 8), P(pi + 9), AGG, co2, N, 1);
    // bbmlp2: FX1 -> FX2 [B,co2,NT]
    pw(FX1, co2, nullptr, 0, P(pi + 10), P(pi + 11), P(pi + 12), FX2, co2, NT, 1);
    // ap2 fc + pool -> WSb
    ap(AGG, co2, FX2, co2, neigh, P(pi + 13), WSb, N);
    // ap2 mlp -> BBo [B,co,N]
    pw(WSb, co, nullptr, 0, P(pi + 14), P(pi + 15), P(pi + 16), BBo, co, N, 1);
    // mlp2 -> M2 [B,2co,N];  mlp3 shortcut -> M3
    pw(BBo, co, nullptr, 0, P(pi + 17), P(pi + 18), P(pi + 19), M2, 2 * co, N, 1);
    pw(fin, cin, nullptr, 0, P(pi + 20), P(pi + 21), P(pi + 22), M3, 2 * co, N, 1);
    // residual: fe = lrelu(M2 + M3)  (enc0's fe persists for the decoder)
    float* fe = (i == 0) ? ENC0 : M2;
    {
      const int n = B * 2 * co * N;
      add_lrelu_k<<<(n + 255) / 256, 256, 0, stream>>>(M2, M3, fe, n);
    }
    // random-sample max over K -> FS[i+1] [B,2co,NS[i+1]]
    {
      const int M = NS[i + 1], d = 2 * co;
      dim3 grid((M + 255) / 256, d, B);
      sample_max_k<<<grid, 256, 0, stream>>>(fe, sub, FS[i + 1], d, N, M);
    }
    fin = FS[i + 1];
    cin = 2 * co;
  }

  // dec0: FS[4] [B,512,160] -> PP2
  pw(FS[4], 512, nullptr, 0, P(95), P(96), P(97), PP2, 512, NS[4], 1);

  const float* encp[4] = {FS[3], FS[2], FS[1], ENC0};
  const int    encc[4] = {256, 128, 32, 32};
  const int    dcout[4] = {256, 128, 32, 32};
  int fd = 512;                                    // channels of current f (PP2)
  for (int j = 0; j < 4; ++j) {
    const int M  = NS[3 - j];                      // upsample target points
    const int Nq = NS[4 - j];                      // source points
    const int* ip = (const int*)I(4 + 4 * (3 - j)); // interp_{3-j}
    {
      dim3 grid((M + 255) / 256, fd, B);
      interp_k<<<grid, 256, 0, stream>>>(PP2, ip, FI, fd, Nq, M);
    }
    const int w0 = 98 + 3 * j;                     // dec{j+1}_{W,s,b}
    pw(encp[j], encc[j], FI, fd, P(w0), P(w0 + 1), P(w0 + 2), PP2, dcout[j], M, 1);
    fd = dcout[j];
  }

  // fc1 (32->64), fc2 (64->32), fc3 (32->13, no BN / no activation) -> d_out
  pw(PP2, 32, nullptr, 0, P(110), P(111), P(112), PP1, 64, NS[0], 1);
  pw(PP1, 64, nullptr, 0, P(113), P(114), P(115), PP2, 32, NS[0], 1);
  pw(PP2, 32, nullptr, 0, P(116), nullptr, nullptr, (float*)d_out, 13, NS[0], 0);
}
